// DeformableFeatureNet_57131654971633
// MI455X (gfx1250) — matrix operand order repaired and checked by reading, not compiled, not run on hardware
//
#include <hip/hip_runtime.h>
#include <cstdint>
#include <cstddef>

typedef __attribute__((ext_vector_type(16))) _Float16 v16h;
typedef __attribute__((ext_vector_type(8)))  float    v8f;
typedef __attribute__((ext_vector_type(4)))  unsigned int v4u;
typedef __attribute__((ext_vector_type(8)))  int      v8i;
typedef __attribute__((ext_vector_type(4)))  int      v4i;

#define EPSV 1e-5f
#define LDS_W_MAX (16 * 448)   // 16 rows x max Kpad(416)+slack, f16

// ---------------------------------------------------------------------------
// Kernels
// ---------------------------------------------------------------------------

__global__ void pack_w_kernel(const float* __restrict__ w, _Float16* __restrict__ wp,
                              int Kreal, int Kpad, int n) {
  int i = blockIdx.x * blockDim.x + threadIdx.x;
  if (i >= n) return;
  int o = i / Kpad, k = i - o * Kpad;
  wp[i] = (k < Kreal) ? (_Float16)w[(size_t)o * Kreal + k] : (_Float16)0.0f;
}

// Branchless guarded sample: clamped address (always in-bounds load),
// validity as a 0/1 multiplier -> no exec-mask branching per tap.
__device__ __forceinline__ float samp(const float* __restrict__ x, int c,
                                      int H, int W, int iy, int ix) {
  int yc = min(max(iy, 0), H - 1);
  int xc = min(max(ix, 0), W - 1);
  float v = x[((size_t)c * H + yc) * W + xc];
  float ok = (iy == yc && ix == xc) ? 1.f : 0.f;
  return v * ok;
}

// TDM: stage a 2-D f16 tile (tile_d0 elems/row x tile_d1 rows) into LDS off 0.
// D# bit packing per CDNA5 ISA 8.3/8.4 (group0 128b, group1 256b).
// Rows beyond tensor_d1 are OOB -> zero-filled in LDS.
__device__ __forceinline__ void tdm_load_tile_f16(unsigned lds_off, const void* gaddr,
                                                  unsigned tile_d0, unsigned tile_d1,
                                                  unsigned tensor_d0, unsigned tensor_d1,
                                                  unsigned stride0_elems) {
#if defined(__gfx1250__) && __has_builtin(__builtin_amdgcn_tensor_load_to_lds)
  unsigned long long ga = (unsigned long long)gaddr;
  v4u g0;
  g0.x = 1u;                                            // count=1, user mode
  g0.y = lds_off;                                       // lds_addr (bytes)
  g0.z = (unsigned)(ga & 0xffffffffu);                  // global_addr[31:0]
  g0.w = (unsigned)((ga >> 32) & 0x01ffffffu) | (2u << 30); // addr[56:32], type=2
  v8i g1;
  g1[0] = (int)(1u << 16);                              // data_size=1 (2 bytes)
  g1[1] = (int)((tensor_d0 & 0xffffu) << 16);           // tensor_dim0[15:0]
  g1[2] = (int)((tensor_d0 >> 16) & 0xffffu) |
          (int)((tensor_d1 & 0xffffu) << 16);           // dim0[31:16] | dim1[15:0]
  g1[3] = (int)((tensor_d1 >> 16) & 0xffffu) |
          (int)((tile_d0 & 0xffffu) << 16);             // dim1[31:16] | tile_dim0
  g1[4] = (int)(tile_d1 & 0xffffu);                     // tile_dim1 (tile_dim2=0)
  g1[5] = (int)stride0_elems;                           // tensor_dim0_stride[31:0]
  g1[6] = 0;                                            // stride0 hi / stride1 lo
  g1[7] = 0;
  v4i z4 = {0, 0, 0, 0};
  v8i z8 = {0, 0, 0, 0, 0, 0, 0, 0};
  __builtin_amdgcn_tensor_load_to_lds(g0, g1, z4, z4, z8, 0);
  __builtin_amdgcn_s_wait_tensorcnt(0);
#else
  (void)lds_off; (void)gaddr; (void)tile_d0; (void)tile_d1;
  (void)tensor_d0; (void)tensor_d1; (void)stride0_elems;
#endif
}

// Implicit-GEMM conv (DEF=0) / deformable conv (DEF=1) with WMMA.
// out[n][m] = sum_k patch[m][k] * W[n][k]. One wave = 16(M)x16(N) tile,
// 4 waves/block; the block's 16-row weight tile lives in LDS (TDM-staged).
template <int KK, bool DEF>
__global__ void __launch_bounds__(128)
gemm_conv_kernel(const float* __restrict__ x,
                 const float* __restrict__ dyp, long dyks,
                 const float* __restrict__ dxp, long dxks,
                 const float* __restrict__ maskp,
                 const _Float16* __restrict__ wp, const float* __restrict__ bias,
                 float* __restrict__ out,
                 int H, int W, int Cout, int Ho, int Wo,
                 int stride, int pad, int Kreal, int Kpad) {
  constexpr int K2 = KK * KK;
  __shared__ _Float16 wlds[LDS_W_MAX];          // first (only) LDS object -> offset 0
  const int M = Ho * Wo;
  const int nbase = blockIdx.y * 16;

  // ---- stage this block's 16 x Kpad weight tile into LDS (before any exit) ----
#if defined(__gfx1250__) && __has_builtin(__builtin_amdgcn_tensor_load_to_lds)
  if (threadIdx.x < 32) {
    tdm_load_tile_f16(0u, wp + (size_t)nbase * Kpad,
                      (unsigned)Kpad, 16u,
                      (unsigned)Kpad, (unsigned)(Cout - nbase),
                      (unsigned)Kpad);
  }
#else
  for (int r = 0; r < 16; ++r)
    for (int c = threadIdx.x; c < Kpad; c += 128)
      wlds[r * Kpad + c] = (nbase + r < Cout)
                               ? wp[(size_t)(nbase + r) * Kpad + c]
                               : (_Float16)0.0f;
#endif
  __syncthreads();

  const int lane = threadIdx.x & 31;
  const int wave = threadIdx.x >> 5;
  const int mtiles = (M + 15) >> 4;
  const int mtile = blockIdx.x * 4 + wave;
  if (mtile >= mtiles) return;                  // wave-uniform; barrier already passed
  const int mrow = mtile * 16 + (lane & 15);
  const bool mvalid = mrow < M;
  const int mr = mvalid ? mrow : 0;             // clamped row for address formation
  const float mscale = mvalid ? 1.f : 0.f;
  const int ncol = nbase + (lane & 15);
  const int khalf  = (lane >> 4) << 3;          // A fragment K sub-block
  const int kbhalf = (lane >> 4) << 4;          // unused; B fragment also indexes with khalf
  const int oy = mr / Wo;                       // once per thread, outside K-loop
  const int ox = mr - oy * Wo;
  const _Float16* brow = &wlds[(size_t)(lane & 15) * Kpad];

  v8f acc = {};
  for (int k0 = 0; k0 < Kpad; k0 += 32) {
    v16h a, b;
#pragma unroll
    for (int e = 0; e < 16; ++e) {
      int k = k0 + khalf + (e & 7) + ((e >> 3) << 4);
      int kc = (k < Kreal) ? k : 0;             // clamped k for address formation
      float kscale = (k < Kreal) ? mscale : 0.f;
      int c = kc / K2, kk = kc - c * K2;        // compile-time divisors
      int ky = kk / KK, kx = kk - ky * KK;
      float v;
      if (DEF) {
        float py = (float)(oy * stride - pad + ky) + dyp[(size_t)kk * dyks + mr];
        float px = (float)(ox * stride - pad + kx) + dxp[(size_t)kk * dxks + mr];
        float y0f = floorf(py), x0f = floorf(px);
        float wy = py - y0f, wx = px - x0f;
        int y0 = (int)y0f, x0 = (int)x0f;
        float v00 = samp(x, c, H, W, y0,     x0);
        float v01 = samp(x, c, H, W, y0,     x0 + 1);
        float v10 = samp(x, c, H, W, y0 + 1, x0);
        float v11 = samp(x, c, H, W, y0 + 1, x0 + 1);
        v = ((1.f - wy) * ((1.f - wx) * v00 + wx * v01) +
             wy         * ((1.f - wx) * v10 + wx * v11)) *
            maskp[(size_t)kk * M + mr];
      } else {
        int iy = oy * stride - pad + ky;
        int ix = ox * stride - pad + kx;
        v = samp(x, c, H, W, iy, ix);           // branchless clamp+mask
      }
      a[e] = (_Float16)(v * kscale);
    }
#pragma unroll
    for (int e = 0; e < 16; ++e)                // LDS B fragment (zero-padded rows)
      b[e] = brow[k0 + khalf + (e & 7) + ((e >> 3) << 4)];
    acc = __builtin_amdgcn_wmma_f32_16x16x32_f16(false, a, false, b,
                                                 (short)0, acc, false, false);
  }
  if (ncol < Cout) {
    float bv = bias ? bias[ncol] : 0.f;
#pragma unroll
    for (int r = 0; r < 8; ++r) {
      int m = mtile * 16 + r + ((lane >> 4) << 3);
      if (m < M) out[(size_t)ncol * M + m] = acc[r] + bv;
    }
  }
}

__global__ void sigmoid_scale_kernel(float* __restrict__ x, long n, float scale) {
  long i = (long)blockIdx.x * blockDim.x + threadIdx.x;
  if (i < n) x[i] = scale * (1.0f / (1.0f + __expf(-x[i])));
}

__global__ void norm_stats_kernel(const float* __restrict__ x,
                                  float* __restrict__ mean, float* __restrict__ istd,
                                  int cpg, int M, float eps) {
  const int g = blockIdx.x;
  const float* xs = x + (size_t)g * cpg * M;
  const long n = (long)cpg * M;
  float s = 0.f, s2 = 0.f;
  for (long i = threadIdx.x; i < n; i += blockDim.x) {
    float v = xs[i]; s += v; s2 += v * v;
  }
  __shared__ float rs[256], rq[256];
  int t = threadIdx.x;
  rs[t] = s; rq[t] = s2;
  __syncthreads();
  for (int off = 128; off > 0; off >>= 1) {
    if (t < off) { rs[t] += rs[t + off]; rq[t] += rq[t + off]; }
    __syncthreads();
  }
  if (t == 0) {
    float mu  = rs[0] / (float)n;
    float var = rq[0] / (float)n - mu * mu;
    mean[g] = mu;
    istd[g] = rsqrtf(fmaxf(var, 0.f) + eps);
  }
}

__global__ void norm_apply_kernel(float* __restrict__ y,
                                  const float* __restrict__ mean,
                                  const float* __restrict__ istd,
                                  const float* __restrict__ g,
                                  const float* __restrict__ b,
                                  int C, int M, int cpg, int relu) {
  long i = (long)blockIdx.x * blockDim.x + threadIdx.x;
  if (i >= (long)C * M) return;
  int c  = (int)(i / M);
  int gg = c / cpg;
  float v = (y[i] - mean[gg]) * istd[gg] * g[c] + b[c];
  if (relu) v = fmaxf(v, 0.f);
  y[i] = v;
}

__global__ void up2_add_kernel(float* __restrict__ out, const float* __restrict__ a,
                               int C, int Hs, int Ws) {
  const long n = (long)C * 4 * Hs * Ws;
  long i = (long)blockIdx.x * blockDim.x + threadIdx.x;
  if (i >= n) return;
  const int  Wl = 2 * Ws;
  const long hw = (long)4 * Hs * Ws;
  int  c = (int)(i / hw);
  long r = i - (long)c * hw;
  int yy = (int)(r / Wl);
  int xx = (int)(r - (long)yy * Wl);
  out[i] += a[((size_t)c * Hs + (yy >> 1)) * Ws + (xx >> 1)];
}

// ---------------------------------------------------------------------------
// Host orchestration
// ---------------------------------------------------------------------------

namespace {

struct Arena {
  char* base; size_t off, cap;
  void* alloc(size_t bytes) {
    size_t o = (off + 255) & ~(size_t)255;
    off = o + bytes;
    return (void*)(base + o);
  }
  float* allocf(size_t elems) { return (float*)alloc(elems * sizeof(float)); }
};

struct DcvP { const float *b2,*g,*mb,*mw,*ob,*ow,*w; };
struct DcnP { const float *b,*omb,*omw,*w; };
struct HeadP { DcnP d1,d2,d3; DcvP dcv; const float *gn1b,*gn1g,*gn2b,*gn2g; };

static unsigned gdiv(long n, int b) { return (unsigned)((n + b - 1) / b); }

static void run_gemm(hipStream_t s, const float* x,
                     const float* dy, long dyks, const float* dx, long dxks,
                     const float* mask, const _Float16* wp, const float* bias,
                     float* out, int H, int W, int Cout, int Ho, int Wo,
                     int Kk, int stride, int pad, int Kreal, int Kpad) {
  int M = Ho * Wo;
  dim3 grid(gdiv((M + 15) / 16, 4), (unsigned)((Cout + 15) / 16));
#define ARGS x, dy, dyks, dx, dxks, mask, wp, bias, out, H, W, Cout, Ho, Wo, \
             stride, pad, Kreal, Kpad
  if (mask) {
    if (Kk == 3)      gemm_conv_kernel<3, true ><<<grid, 128, 0, s>>>(ARGS);
    else if (Kk == 5) gemm_conv_kernel<5, true ><<<grid, 128, 0, s>>>(ARGS);
  } else {
    if (Kk == 1)      gemm_conv_kernel<1, false><<<grid, 128, 0, s>>>(ARGS);
    else if (Kk == 3) gemm_conv_kernel<3, false><<<grid, 128, 0, s>>>(ARGS);
    else if (Kk == 5) gemm_conv_kernel<5, false><<<grid, 128, 0, s>>>(ARGS);
  }
#undef ARGS
}

static _Float16* pack_w(hipStream_t s, Arena& A, const float* w, int Cout,
                        int Kreal, int Kpad) {
  _Float16* wp = (_Float16*)A.alloc((size_t)Cout * Kpad * sizeof(_Float16));
  int np = Cout * Kpad;
  pack_w_kernel<<<gdiv(np, 256), 256, 0, s>>>(w, wp, Kreal, Kpad, np);
  return wp;
}

static void launch_conv(hipStream_t s, Arena& A, const float* x, int Cin, int H, int W,
                        const float* w, const float* bias, int Cout, int Kk,
                        int stride, int pad, float* out, int Ho, int Wo) {
  int Kreal = Cin * Kk * Kk, Kpad = (Kreal + 31) & ~31;
  size_t mark = A.off;
  _Float16* wp = pack_w(s, A, w, Cout, Kreal, Kpad);
  run_gemm(s, x, nullptr, 0, nullptr, 0, nullptr, wp, bias, out,
           H, W, Cout, Ho, Wo, Kk, stride, pad, Kreal, Kpad);
  A.off = mark;  // stream-ordered reuse
}

static void launch_dconv(hipStream_t s, Arena& A, const float* x, int Cin, int H, int W,
                         const float* dy, long dyks, const float* dx, long dxks,
                         const float* mask, const float* w, const float* bias,
                         int Cout, int Kk, int stride, int pad,
                         float* out, int Ho, int Wo) {
  int Kreal = Cin * Kk * Kk, Kpad = (Kreal + 31) & ~31;
  size_t mark = A.off;
  _Float16* wp = pack_w(s, A, w, Cout, Kreal, Kpad);
  run_gemm(s, x, dy, dyks, dx, dxks, mask, wp, bias, out,
           H, W, Cout, Ho, Wo, Kk, stride, pad, Kreal, Kpad);
  A.off = mark;
}

static void launch_norm(hipStream_t s, Arena& A, float* y, int C, int M,
                        const float* g, const float* b, int groups, int relu) {
  size_t mark = A.off;
  float* stats = A.allocf((size_t)2 * groups);
  norm_stats_kernel<<<groups, 256, 0, s>>>(y, stats, stats + groups, C / groups, M, EPSV);
  long n = (long)C * M;
  norm_apply_kernel<<<gdiv(n, 256), 256, 0, s>>>(y, stats, stats + groups, g, b,
                                                 C, M, C / groups, relu);
  A.off = mark;
}

static void dcv_block(hipStream_t s, Arena& A, const float* x, int Cin, int H, int W,
                      const DcvP& p, int Cout, int Kk, int stride, int pad,
                      float* out, int& Ho, int& Wo) {
  Ho = (H + 2 * pad - Kk) / stride + 1;
  Wo = (W + 2 * pad - Kk) / stride + 1;
  const int M = Ho * Wo, K2 = Kk * Kk;
  size_t mark = A.off;
  float* off = A.allocf((size_t)2 * K2 * M);
  launch_conv(s, A, x, Cin, H, W, p.ow, p.ob, 2 * K2, Kk, stride, pad, off, Ho, Wo);
  float* mod = A.allocf((size_t)K2 * M);
  launch_conv(s, A, x, Cin, H, W, p.mw, p.mb, K2, Kk, stride, pad, mod, Ho, Wo);
  long nm = (long)K2 * M;
  sigmoid_scale_kernel<<<gdiv(nm, 256), 256, 0, s>>>(mod, nm, 2.0f);
  // off is (K2,2,Ho,Wo): dy = chan 2*kk (stride 2M), dx = chan 2*kk+1
  launch_dconv(s, A, x, Cin, H, W, off, 2L * M, off + M, 2L * M, mod,
               p.w, nullptr, Cout, Kk, stride, pad, out, Ho, Wo);
  launch_norm(s, A, out, Cout, M, p.g, p.b2, 4, 1);
  A.off = mark;
}

static void dcn_block(hipStream_t s, Arena& A, const float* x, int Cin, int H, int W,
                      const DcnP& p, int Cout, float* out) {
  const int M = H * W;
  size_t mark = A.off;
  float* om = A.allocf((size_t)27 * M);
  launch_conv(s, A, x, Cin, H, W, p.omw, p.omb, 27, 3, 1, 1, om, H, W);
  long nm = 9L * M;
  sigmoid_scale_kernel<<<gdiv(nm, 256), 256, 0, s>>>(om + 18L * M, nm, 1.0f);
  launch_dconv(s, A, x, Cin, H, W, om, (long)M, om + 9L * M, (long)M, om + 18L * M,
               p.w, p.b, Cout, 3, 1, 1, out, H, W);
  A.off = mark;
}

static void out_head(hipStream_t s, Arena& A, const float* x, int H, int W,
                     const HeadP& p, int CoutLast, float* out) {
  const int M = H * W;
  size_t mark = A.off;
  float* y  = A.allocf((size_t)32 * M);
  int Ho, Wo;
  dcv_block(s, A, x, 32, H, W, p.dcv, 32, 3, 1, 1, y, Ho, Wo);
  float* y2 = A.allocf((size_t)32 * M);
  dcn_block(s, A, y, 32, H, W, p.d1, 32, y2);
  launch_norm(s, A, y2, 32, M, p.gn1g, p.gn1b, 4, 1);
  dcn_block(s, A, y2, 32, H, W, p.d2, 32, y);
  launch_norm(s, A, y, 32, M, p.gn2g, p.gn2b, 4, 1);
  dcn_block(s, A, y, 32, H, W, p.d3, CoutLast, out);
  A.off = mark;
}

} // namespace

extern "C" void kernel_launch(void* const* d_in, const int* in_sizes, int n_in,
                              void* d_out, int out_size, void* d_ws, size_t ws_size,
                              hipStream_t stream) {
  (void)in_sizes; (void)n_in; (void)out_size;
  // d_in follows jax pytree flatten order: dicts alphabetical, lists in order.
  int idx = 0;
  auto nxt = [&]() { return (const float*)d_in[idx++]; };
  auto rd_dcv = [&]() { DcvP p; p.b2=nxt(); p.g=nxt(); p.mb=nxt(); p.mw=nxt();
                        p.ob=nxt(); p.ow=nxt(); p.w=nxt(); return p; };
  auto rd_dcn = [&]() { DcnP p; p.b=nxt(); p.omb=nxt(); p.omw=nxt(); p.w=nxt(); return p; };
  auto rd_head = [&]() { HeadP h; h.d1=rd_dcn(); h.d2=rd_dcn(); h.d3=rd_dcn();
                         h.dcv=rd_dcv(); h.gn1b=nxt(); h.gn1g=nxt();
                         h.gn2b=nxt(); h.gn2g=nxt(); return h; };

  DcvP conv0_0 = rd_dcv(), conv0_1 = rd_dcv();
  DcvP conv1_0 = rd_dcv(), conv1_1 = rd_dcv(), conv1_2 = rd_dcv();
  DcvP conv2_0 = rd_dcv(), conv2_1 = rd_dcv(), conv2_2 = rd_dcv();
  const float* inner1_b = nxt(); const float* inner1_w = nxt();
  const float* inner2_b = nxt(); const float* inner2_w = nxt();
  const float* o1_bnb = nxt(); const float* o1_bng = nxt(); const float* o1_c1w = nxt();
  DcnP o1_d1 = rd_dcn(), o1_d2 = rd_dcn(), o1_d3 = rd_dcn();
  const float* o1_gn1b = nxt(); const float* o1_gn1g = nxt();
  const float* o1_gn2b = nxt(); const float* o1_gn2g = nxt();
  HeadP o2 = rd_head(), o3 = rd_head();
  const float* x = nxt();

  Arena A{(char*)d_ws, 0, ws_size};
  const int H0 = 512, W0 = 640, M0 = H0 * W0;
  const int H1 = 256, W1 = 320, M1 = H1 * W1;
  const int H2 = 128, W2 = 160, M2 = H2 * W2;
  int Ho, Wo;

  float* c0     = A.allocf((size_t)8  * M0);
  float* c1     = A.allocf((size_t)16 * M1);
  float* c2     = A.allocf((size_t)32 * M2);
  float* intra  = A.allocf((size_t)32 * M1);
  float* intra2 = A.allocf((size_t)32 * M0);

  { size_t mark = A.off;
    float* t = A.allocf((size_t)8 * M0);
    dcv_block(stream, A, x, 3, H0, W0, conv0_0, 8, 3, 1, 1, t,  Ho, Wo);
    dcv_block(stream, A, t, 8, H0, W0, conv0_1, 8, 3, 1, 1, c0, Ho, Wo);
    A.off = mark; }
  { size_t mark = A.off;
    float* t1 = A.allocf((size_t)16 * M1);
    float* t2 = A.allocf((size_t)16 * M1);
    dcv_block(stream, A, c0, 8,  H0, W0, conv1_0, 16, 5, 2, 2, t1, Ho, Wo);
    dcv_block(stream, A, t1, 16, H1, W1, conv1_1, 16, 3, 1, 1, t2, Ho, Wo);
    dcv_block(stream, A, t2, 16, H1, W1, conv1_2, 16, 3, 1, 1, c1, Ho, Wo);
    A.off = mark; }
  { size_t mark = A.off;
    float* t1 = A.allocf((size_t)32 * M2);
    float* t2 = A.allocf((size_t)32 * M2);
    dcv_block(stream, A, c1, 16, H1, W1, conv2_0, 32, 5, 2, 2, t1, Ho, Wo);
    dcv_block(stream, A, t1, 32, H2, W2, conv2_1, 32, 3, 1, 1, t2, Ho, Wo);
    dcv_block(stream, A, t2, 32, H2, W2, conv2_2, 32, 3, 1, 1, c2, Ho, Wo);
    A.off = mark; }

  float* outp   = (float*)d_out;
  float* stage1 = outp;                          // (32,128,160)
  float* stage2 = stage1 + (size_t)32 * M2;      // (16,256,320)
  float* stage3 = stage2 + (size_t)16 * M1;      // (8,512,640)

  { size_t mark = A.off;
    float* y  = A.allocf((size_t)32 * M2);
    float* y2 = A.allocf((size_t)32 * M2);
    launch_conv(stream, A, c2, 32, H2, W2, o1_c1w, nullptr, 32, 1, 1, 0, y, H2, W2);
    launch_norm(stream, A, y, 32, M2, o1_bng, o1_bnb, 32, 1);  // batch norm, B=1
    dcn_block(stream, A, y, 32, H2, W2, o1_d1, 32, y2);
    launch_norm(stream, A, y2, 32, M2, o1_gn1g, o1_gn1b, 4, 1);
    dcn_block(stream, A, y2, 32, H2, W2, o1_d2, 32, y);
    launch_norm(stream, A, y, 32, M2, o1_gn2g, o1_gn2b, 4, 1);
    dcn_block(stream, A, y, 32, H2, W2, o1_d3, 32, stage1);
    A.off = mark; }

  launch_conv(stream, A, c1, 16, H1, W1, inner1_w, inner1_b, 32, 1, 1, 0, intra, H1, W1);
  { long n = (long)32 * M1;
    up2_add_kernel<<<gdiv(n, 256), 256, 0, stream>>>(intra, c2, 32, H2, W2); }
  out_head(stream, A, intra, H1, W1, o2, 16, stage2);

  launch_conv(stream, A, c0, 8, H0, W0, inner2_w, inner2_b, 32, 1, 1, 0, intra2, H0, W0);
  { long n = (long)32 * M0;
    up2_add_kernel<<<gdiv(n, 256), 256, 0, stream>>>(intra2, intra, 32, H1, W1); }
  out_head(stream, A, intra2, H0, W0, o3, 8, stage3);
}
